// equivariant_layer_10325101379770
// MI455X (gfx1250) — compile-verified, matrix operand order checked by reading, not run on hardware
//
#include <hip/hip_runtime.h>
#include <math.h>

typedef __attribute__((ext_vector_type(16))) _Float16 v16h;
typedef __attribute__((ext_vector_type(8)))  float    v8f;

#define HDIM   128        // hidden width per MLP (H)
#define KSTEPS 9          // ceil(257 / 32) k-steps of the WMMA
#define NTILES 16         // 256 combined hidden units / 16
#define BFRAG_ELEMS (NTILES * KSTEPS * 32 * 16)   // packed f16 B fragments
#define SMEM_BYTES  (BFRAG_ELEMS * 2)             // 147456 bytes of LDS

// ---------------------------------------------------------------- zero scratch
__global__ void zero_f32_kernel(float* __restrict__ p, int n) {
    int i = blockIdx.x * blockDim.x + threadIdx.x;
    if (i < n) p[i] = 0.0f;
}

// ------------------------------------------------- pack weights into B layout
// Fragment layout (per (ntile c, kstep t)): lane holds column n = 16c + lane%16,
// 16 f16 K-values packed as K-offsets {b0..b0+7, 16+b0..16+b0+7}, b0 = 8*(lane>=16).
__global__ void prep_b_kernel(const float* __restrict__ w1_0,
                              const float* __restrict__ w2_0,
                              _Float16* __restrict__ bfrag) {
    int idx = blockIdx.x * blockDim.x + threadIdx.x;
    if (idx >= BFRAG_ELEMS) return;
    int i    = idx & 15;
    int lane = (idx >> 4) & 31;
    int ct   = idx >> 9;          // c*KSTEPS + t
    int t    = ct % KSTEPS;
    int c    = ct / KSTEPS;
    int n    = c * 16 + (lane & 15);
    int b0   = (lane & 16) ? 8 : 0;
    int koff = (i < 8) ? (b0 + i) : (16 + b0 + (i - 8));
    int k    = t * 32 + koff;
    float v  = 0.0f;
    if (k < 2 * HDIM + 1)
        v = (n < HDIM) ? w1_0[k * HDIM + n] : w2_0[k * HDIM + (n - HDIM)];
    bfrag[idx] = (_Float16)v;
}

// ----------------------------------------------------- A fragment construction
__device__ __forceinline__ void build_afrag(const float* __restrict__ x,
                                            int src, int dst, float dist,
                                            int b0, v16h* afrag) {
    const float* __restrict__ rs = x + (size_t)src * HDIM;
    const float* __restrict__ rd = x + (size_t)dst * HDIM;
#pragma unroll
    for (int t = 0; t < 8; ++t) {
        const float* r = (t < 4) ? rs : rd;
        const int k0 = (t & 3) * 32 + b0;
        v16h a;
#pragma unroll
        for (int i = 0; i < 8; ++i) {
            a[i]     = (_Float16)r[k0 + i];
            a[i + 8] = (_Float16)r[k0 + 16 + i];
        }
        afrag[t] = a;
    }
    v16h a8;
#pragma unroll
    for (int i = 0; i < 16; ++i) a8[i] = (_Float16)0.0f;
    if (b0 == 0) a8[0] = (_Float16)dist;              // K = 256 is the dist column
    afrag[8] = a8;
}

// --------------------------------------------------------- edge MLP via WMMA
// Each wave owns a 32-edge group (two 16-row M-tiles sharing every B fragment).
// B fragments live in LDS (filled once per block); B reads are software-
// pipelined with an explicit double buffer so ds latency overlaps WMMA issue.
__global__ __launch_bounds__(256)
void edge_mlp_wmma_kernel(const float* __restrict__ x,
                          const float* __restrict__ pos,
                          const long long* __restrict__ ei,
                          const _Float16* __restrict__ bfrag,
                          const float* __restrict__ b1_0,
                          const float* __restrict__ w1_1,
                          const float* __restrict__ b1_1,
                          const float* __restrict__ b2_0,
                          const float* __restrict__ w2_1,
                          const float* __restrict__ b2_1,
                          float* __restrict__ v1,
                          float* __restrict__ v2,
                          int E, int ngroups) {
    extern __shared__ char smem_raw[];
    _Float16* sB = (_Float16*)smem_raw;

    // ---- cooperative fill of B fragments into LDS (147456 B, 16B vectors)
    {
        const uint4* gsrc = (const uint4*)bfrag;
        uint4* ldst = (uint4*)smem_raw;
        for (int i = threadIdx.x; i < SMEM_BYTES / 16; i += blockDim.x)
            ldst[i] = gsrc[i];
    }
    __syncthreads();

    const int lane = threadIdx.x & 31;
    const int wave = threadIdx.x >> 5;
    const int nl   = lane & 15;
    const int b0   = (lane & 16) ? 8 : 0;
    const int waveGlobal = blockIdx.x * 8 + wave;
    const int nWavesTot  = gridDim.x * 8;

    for (int g = waveGlobal; g < ngroups; g += nWavesTot) {   // wave-uniform loop
        // ---- gather both M-tiles' edge data
        const int e0row = g * 32 + nl;
        const int e1row = g * 32 + 16 + nl;
        const bool valid0 = (e0row < E), valid1 = (e1row < E);
        const int e0 = valid0 ? e0row : (E - 1);
        const int e1 = valid1 ? e1row : (E - 1);
        const int srcA = (int)ei[e0], dstA = (int)ei[(size_t)E + e0];
        const int srcB = (int)ei[e1], dstB = (int)ei[(size_t)E + e1];
        const float rxA = pos[srcA*3+0]-pos[dstA*3+0];
        const float ryA = pos[srcA*3+1]-pos[dstA*3+1];
        const float rzA = pos[srcA*3+2]-pos[dstA*3+2];
        const float rxB = pos[srcB*3+0]-pos[dstB*3+0];
        const float ryB = pos[srcB*3+1]-pos[dstB*3+1];
        const float rzB = pos[srcB*3+2]-pos[dstB*3+2];
        const float distA = sqrtf(rxA*rxA + ryA*ryA + rzA*rzA);
        const float distB = sqrtf(rxB*rxB + ryB*ryB + rzB*rzB);

        v16h afragA[KSTEPS], afragB[KSTEPS];
        build_afrag(x, srcA, dstA, distA, b0, afragA);
        build_afrag(x, srcB, dstB, distB, b0, afragB);

        // ---- GEMM: c rolled (register control), t unrolled, B shared by 2 tiles
        v8f acc1A = {}, acc2A = {}, acc1B = {}, acc2B = {};
#pragma unroll 1
        for (int c = 0; c < NTILES; ++c) {
            const float bias = (c < 8) ? b1_0[c*16 + nl] : b2_0[(c-8)*16 + nl];
            const float wt   = (c < 8) ? w1_1[c*16 + nl] : w2_1[(c-8)*16 + nl];
            const _Float16* bp = sB + ((size_t)(c * KSTEPS) * 32 + lane) * 16;
            v8f d0 = {}, d1 = {};
            v16h bcur = *(const v16h*)bp;             // prologue load (t = 0)
#pragma unroll
            for (int t = 0; t < KSTEPS; ++t) {
                // issue next B load BEFORE consuming current one
                v16h bnext = (t + 1 < KSTEPS)
                               ? *(const v16h*)(bp + (size_t)(t + 1) * 32 * 16)
                               : bcur;
                d0 = __builtin_amdgcn_wmma_f32_16x16x32_f16(
                        false, afragA[t], false, bcur, (short)0, d0, false, false);
                d1 = __builtin_amdgcn_wmma_f32_16x16x32_f16(
                        false, afragB[t], false, bcur, (short)0, d1, false, false);
                bcur = bnext;
            }
#pragma unroll
            for (int i = 0; i < 8; ++i) {
                float h0 = d0[i] + bias; h0 = (h0 > 0.0f) ? h0 : 0.0f;
                float h1 = d1[i] + bias; h1 = (h1 > 0.0f) ? h1 : 0.0f;
                if (c < 8) { acc1A[i] += h0 * wt; acc1B[i] += h1 * wt; }
                else       { acc2A[i] += h0 * wt; acc2B[i] += h1 * wt; }
            }
        }

        // ---- reduce over hidden (N on lanes): xor-reduce each 16-lane half
#pragma unroll
        for (int off = 1; off < 16; off <<= 1) {
#pragma unroll
            for (int i = 0; i < 8; ++i) {
                acc1A[i] += __shfl_xor(acc1A[i], off, 32);
                acc2A[i] += __shfl_xor(acc2A[i], off, 32);
                acc1B[i] += __shfl_xor(acc1B[i], off, 32);
                acc2B[i] += __shfl_xor(acc2B[i], off, 32);
            }
        }
        // Redistribute: lane m (0..15) <- tile0 edge m; lane 16+m <- tile1 edge m.
        float mes1 = 0.0f, mes2 = 0.0f;
#pragma unroll
        for (int i = 0; i < 8; ++i) {
            float l1 = __shfl(acc1A[i], 0, 32),  h1 = __shfl(acc1A[i], 16, 32);
            float l2 = __shfl(acc2A[i], 0, 32),  h2 = __shfl(acc2A[i], 16, 32);
            float l3 = __shfl(acc1B[i], 0, 32),  h3 = __shfl(acc1B[i], 16, 32);
            float l4 = __shfl(acc2B[i], 0, 32),  h4 = __shfl(acc2B[i], 16, 32);
            if (lane == i)      { mes1 = l1; mes2 = l2; }
            if (lane == i + 8)  { mes1 = h1; mes2 = h2; }
            if (lane == i + 16) { mes1 = l3; mes2 = l4; }
            if (lane == i + 24) { mes1 = h3; mes2 = h4; }
        }

        // ---- per-edge epilogue + atomic scatter (all 32 lanes, one edge each)
        const bool  low  = (lane < 16);
        const bool  vOK  = low ? valid0 : valid1;
        const int   srcE = low ? srcA : srcB;
        const float rx   = low ? rxA : rxB;
        const float ry   = low ? ryA : ryB;
        const float rz   = low ? rzA : rzB;
        const float dist = low ? distA : distB;
        if (vOK) {
            mes1 += b1_1[0];
            mes2 += b2_1[0];
            const float dd = dist * (1.0f / 4.5f);
            const float d2 = dd * dd;
            const float d5 = d2 * d2 * dd;
            const float coe = 1.0f - 21.0f*d5 + 35.0f*d5*dd - 15.0f*d5*d2;
            const float inv = 1.0f / (dist + 1e-12f);
            const float s1 = coe * mes1 * inv;
            const float s2 = coe * mes2 * inv;
            atomicAdd(&v1[srcE*3 + 0], rx * s1);
            atomicAdd(&v1[srcE*3 + 1], ry * s1);
            atomicAdd(&v1[srcE*3 + 2], rz * s1);
            atomicAdd(&v2[srcE*3 + 0], rx * s2);
            atomicAdd(&v2[srcE*3 + 1], ry * s2);
            atomicAdd(&v2[srcE*3 + 2], rz * s2);
        }
    }
}

// ----------------------------------------------------------- Gram-Schmidt out
__global__ void gram_schmidt_kernel(const float* __restrict__ v1,
                                    const float* __restrict__ v2,
                                    float* __restrict__ out, int N) {
    int n = blockIdx.x * blockDim.x + threadIdx.x;
    if (n >= N) return;
    float a0 = v1[n*3+0], a1 = v1[n*3+1], a2 = v1[n*3+2];
    float c0 = v2[n*3+0], c1 = v2[n*3+1], c2 = v2[n*3+2];
    float na = sqrtf(a0*a0 + a1*a1 + a2*a2) + 1e-12f;
    float n10 = a0/na, n11 = a1/na, n12 = a2/na;
    float dp = n10*c0 + n11*c1 + n12*c2;
    float p0 = c0 - dp*n10, p1 = c1 - dp*n11, p2 = c2 - dp*n12;
    float np = sqrtf(p0*p0 + p1*p1 + p2*p2) + 1e-12f;
    float n20 = p0/np, n21 = p1/np, n22 = p2/np;
    float n30 = n11*n22 - n12*n21;
    float n31 = n12*n20 - n10*n22;
    float n32 = n10*n21 - n11*n20;
    float* o = out + (size_t)n * 9;
    o[0] = n10; o[1] = n11; o[2] = n12;
    o[3] = n20; o[4] = n21; o[5] = n22;
    o[6] = n30; o[7] = n31; o[8] = n32;
}

extern "C" void kernel_launch(void* const* d_in, const int* in_sizes, int n_in,
                              void* d_out, int out_size, void* d_ws, size_t ws_size,
                              hipStream_t stream) {
    const float*     x     = (const float*)d_in[0];
    const float*     pos   = (const float*)d_in[1];
    const long long* ei    = (const long long*)d_in[2];
    // d_in[3] = batch (unused)
    const float*     w1_0  = (const float*)d_in[4];
    const float*     b1_0  = (const float*)d_in[5];
    const float*     w1_1  = (const float*)d_in[6];
    const float*     b1_1  = (const float*)d_in[7];
    const float*     w2_0  = (const float*)d_in[8];
    const float*     b2_0  = (const float*)d_in[9];
    const float*     w2_1  = (const float*)d_in[10];
    const float*     b2_1  = (const float*)d_in[11];

    const int N = in_sizes[1] / 3;
    const int E = in_sizes[2] / 2;
    const int ngroups = (E + 31) / 32;                 // 32 edges per wave-group

    // scratch layout: v1 [N*3] f32 | v2 [N*3] f32 | packed B fragments (f16)
    float* v1 = (float*)d_ws;
    float* v2 = v1 + (size_t)N * 3;
    _Float16* bfrag = (_Float16*)(v2 + (size_t)N * 3);

    {   // zero accumulators (atomics accumulate; must re-zero every launch)
        int nz = N * 6;
        zero_f32_kernel<<<(nz + 255) / 256, 256, 0, stream>>>(v1, nz);
    }
    prep_b_kernel<<<(BFRAG_ELEMS + 255) / 256, 256, 0, stream>>>(w1_0, w2_0, bfrag);

    int blocks = (ngroups + 7) / 8;
    if (blocks > 640) blocks = 640;                    // grid-stride; ~5 groups/wave
    if (blocks < 1) blocks = 1;
    edge_mlp_wmma_kernel<<<blocks, 256, SMEM_BYTES, stream>>>(
        x, pos, ei, bfrag, b1_0, w1_1, b1_1, b2_0, w2_1, b2_1, v1, v2, E, ngroups);

    gram_schmidt_kernel<<<(N + 255) / 256, 256, 0, stream>>>(v1, v2, (float*)d_out, N);
}
